// GNN19_27410481283388
// MI455X (gfx1250) — matrix operand order repaired and checked by reading, not compiled, not run on hardware
//
#include <hip/hip_runtime.h>

#define Bsz 16
#define Nn  1500
#define Np  1504          // padded node stride for transposed Wh (32B-aligned rows)
#define FIN1 11
#define HH  3
#define F1c 16
#define F2c 64
#define DATT (HH * F2c)   // 192
#define DH  128

typedef __attribute__((ext_vector_type(16))) _Float16 v16h;
typedef __attribute__((ext_vector_type(8)))  float    v8f;
typedef __attribute__((ext_vector_type(4)))  float    f4v;

// ---------------------------------------------------------------------------
// Kernel 1: per-node feature transform.  Wh stored TRANSPOSED in global
// memory as f16 [B,H,FOUT,Np] so the WMMA B fragments become two
// contiguous 16B loads.  Also emits e_src / e_dst attention scalars.
// ---------------------------------------------------------------------------
template<int FIN_, int FOUT_>
__global__ __launch_bounds__(256)
void feat_kernel(const float* __restrict__ X, const float* __restrict__ W,
                 const float* __restrict__ a_s, const float* __restrict__ a_d,
                 _Float16* __restrict__ WhT, float* __restrict__ es,
                 float* __restrict__ ed)
{
    int t = blockIdx.x * blockDim.x + threadIdx.x;
    if (t >= Bsz * Nn) return;
    int b = t / Nn, n = t % Nn;

    float x[FIN_];
#pragma unroll
    for (int f = 0; f < FIN_; ++f) x[f] = X[(size_t)t * FIN_ + f];

    for (int h = 0; h < HH; ++h) {
        float se = 0.f, sd = 0.f;
        size_t colBase = ((size_t)(b * HH + h) * FOUT_) * Np + n;
        for (int o = 0; o < FOUT_; ++o) {
            float acc = 0.f;
#pragma unroll
            for (int f = 0; f < FIN_; ++f)
                acc += x[f] * W[(h * FIN_ + f) * FOUT_ + o];
            WhT[colBase + (size_t)o * Np] = (_Float16)acc;   // transposed store
            se += acc * a_s[h * FOUT_ + o];
            sd += acc * a_d[h * FOUT_ + o];
        }
        es[((size_t)b * HH + h) * Nn + n] = se;
        ed[((size_t)b * HH + h) * Nn + n] = sd;
    }
    // zero the Np padding columns so boundary B-fragments stay finite
    if (n < Np - Nn) {
        for (int h = 0; h < HH; ++h)
            for (int o = 0; o < FOUT_; ++o)
                WhT[((size_t)(b * HH + h) * FOUT_ + o) * Np + Nn + n] = (_Float16)0.f;
    }
}

// ---------------------------------------------------------------------------
// Kernel 2: FUSED flash-style GAT aggregation.  Single pass over adj:
// online softmax (running row max m, denom l, accumulator rescale by
// exp(m_old-m_new)) wrapped around v_wmma_f32_16x16x32_f16.  One wave per
// 16-row tile; K loop in steps of 32.  No LDS, no block syncs; all operands
// arrive as b128 global loads (adj / e_dst per-lane runs, Wh pre-transposed).
// ---------------------------------------------------------------------------
template<int FOUT_>
__global__ __launch_bounds__(256)
void attn_kernel(const float* __restrict__ adj, const _Float16* __restrict__ WhT,
                 const float* __restrict__ es, const float* __restrict__ ed,
                 float* __restrict__ hout)
{
    constexpr int NT = FOUT_ / 16;
    constexpr int RT = (Nn + 15) / 16;

    const int lane = threadIdx.x & 31;
    const int total = Bsz * HH * RT;
    int task = blockIdx.x * (blockDim.x >> 5) + (threadIdx.x >> 5);
    if (task >= total) task = total - 1;     // duplicate work keeps EXEC full

    const int rt = task % RT;
    const int bh = task / RT;                // b*HH + h
    const int h  = bh % HH;
    const int b  = bh / HH;
    const int i0 = rt * 16;

    const size_t    bhBase = (size_t)bh * Nn;
    const float*    adjB   = adj + (size_t)b * Nn * Nn;
    const _Float16* WhTB   = WhT + (size_t)bh * FOUT_ * Np;
    const float*    edB    = ed + bhBase;

    const int iRow = i0 + (lane & 15);
    const int iRc  = (iRow < Nn) ? iRow : (Nn - 1);   // clamped row (dup rows harmless)
    const float* arow = adjB + (size_t)iRc * Nn;
    const float  esi  = es[bhBase + iRc];

    const int kb0 = (lane < 16) ? 0 : 8;     // 16-bit A 16x32 lane layout
    const int kb2 = (lane < 16) ? 0 : 8;     // 16-bit B 32x16 lane layout
    const int sb  = (lane & 16) >> 1;        // row-group base for shuffles

    float rm = -9e15f;                       // running row max
    float rl = 0.f;                          // running row denom
    v8f acc[NT] = {};

    for (int j0 = 0; j0 < Nn; j0 += 32) {
        float av[16], ev[16];

        if (j0 + 32 <= Nn) {
            // ---- fast path: each lane's 16 (adj, e_dst) values are two
            //      contiguous 8-float runs -> pure b128 loads ------------
            f4v A0 = *(const f4v*)(arow + j0 + kb0);
            f4v A1 = *(const f4v*)(arow + j0 + kb0 + 4);
            f4v A2 = *(const f4v*)(arow + j0 + 16 + kb0);
            f4v A3 = *(const f4v*)(arow + j0 + 16 + kb0 + 4);
            f4v E0 = *(const f4v*)(edB + j0 + kb0);
            f4v E1 = *(const f4v*)(edB + j0 + kb0 + 4);
            f4v E2 = *(const f4v*)(edB + j0 + 16 + kb0);
            f4v E3 = *(const f4v*)(edB + j0 + 16 + kb0 + 4);
#pragma unroll
            for (int q = 0; q < 4; ++q) {
                av[q]      = A0[q];  ev[q]      = E0[q];
                av[4 + q]  = A1[q];  ev[4 + q]  = E1[q];
                av[8 + q]  = A2[q];  ev[8 + q]  = E2[q];
                av[12 + q] = A3[q];  ev[12 + q] = E3[q];
            }
            if (j0 + 64 <= Nn) {
                __builtin_prefetch(arow + j0 + 32, 0, 0);        // next adj run
#pragma unroll
                for (int q = lane; q < FOUT_; q += 32)           // next Wh cols
                    __builtin_prefetch(&WhTB[(size_t)q * Np + j0 + 32], 0, 0);
            }
        } else {
            // ---- boundary: clamped scalar loads + selects --------------
#pragma unroll
            for (int e = 0; e < 16; ++e) {
                int  k   = ((e >= 8) ? 16 : 0) + kb0 + (e & 7);
                int  j   = j0 + k;
                bool jok = (j < Nn);
                int  jc  = jok ? j : (Nn - 1);
                av[e] = jok ? arow[jc] : 0.f;
                ev[e] = jok ? edB[jc]  : 0.f;
            }
        }

        // ---- online softmax over this 32-wide slab ----------------------
        float ex[16];
        float tmax = -9e15f;
#pragma unroll
        for (int e = 0; e < 16; ++e) {
            float xv = esi + ev[e];
            xv = (xv < 0.f) ? 0.2f * xv : xv;          // leaky_relu(0.2)
            xv = (av[e] > 0.f) ? xv : -9e15f;          // adjacency mask
            ex[e] = xv;
            tmax = fmaxf(tmax, xv);
        }
        tmax = fmaxf(tmax, __shfl_xor(tmax, 16, 32));  // join row halves
        float mnew   = fmaxf(rm, tmax);
        float factor = __expf(rm - mnew);

        v16h a;
        float ps = 0.f;
#pragma unroll
        for (int e = 0; e < 16; ++e) {
            float p = (av[e] > 0.f) ? __expf(ex[e] - mnew) : 0.f;
            ps += p;
            a[e] = (_Float16)p;
        }
        ps += __shfl_xor(ps, 16, 32);
        rl = rl * factor + ps;
        bool bumped = (mnew > rm);
        rm = mnew;

        if (__any(bumped)) {                 // uniform; EXEC stays full
            float fr[8];
#pragma unroll
            for (int r = 0; r < 8; ++r) fr[r] = __shfl(factor, sb + r, 32);
#pragma unroll
            for (int ft = 0; ft < NT; ++ft)
#pragma unroll
                for (int r = 0; r < 8; ++r) acc[ft][r] *= fr[r];
        }

        // ---- WMMA accumulate: B fragment = two 16B loads (k runs kb2.., 16+kb2..) ------
#pragma unroll
        for (int ft = 0; ft < NT; ++ft) {
            typedef __attribute__((ext_vector_type(8))) _Float16 v8h;
            const _Float16* bp = &WhTB[(size_t)(ft * 16 + (lane & 15)) * Np + j0 + kb2];
            v8h blo = *(const v8h*)bp;
            v8h bhi = *(const v8h*)(bp + 16);
            v16h bf = __builtin_shufflevector(blo, bhi, 0, 1, 2, 3, 4, 5, 6, 7, 8, 9, 10, 11, 12, 13, 14, 15);
            acc[ft] = __builtin_amdgcn_wmma_f32_16x16x32_f16(
                false, a, false, bf, (short)0, acc[ft], false, false);
        }
    }

    // ---- epilogue: normalize by row denom, ELU, store concatenated heads
    float rli = 1.f / fmaxf(rl, 1e-30f);
    float lr[8];
#pragma unroll
    for (int r = 0; r < 8; ++r) lr[r] = __shfl(rli, sb + r, 32);

    const int col  = lane & 15;
    const int mAdd = sb;                     // 0 or 8
#pragma unroll
    for (int ft = 0; ft < NT; ++ft)
#pragma unroll
        for (int r = 0; r < 8; ++r) {
            int i = i0 + r + mAdd;
            if (i < Nn) {
                float x = acc[ft][r] * lr[r];
                x = (x > 0.f) ? x : (__expf(x) - 1.f);   // ELU
                hout[((size_t)b * Nn + i) * (HH * FOUT_) + h * FOUT_ + ft * 16 + col] = x;
            }
        }
}

// ---------------------------------------------------------------------------
// Kernel 3: self-attention pooling + Dense(192->128, relu) + Dense(128->1).
// One block per batch element.  Softmax weights parked in LDS; pooled rep
// computed as a coalesced column reduction (thread f owns feature f).
// ---------------------------------------------------------------------------
__global__ __launch_bounds__(256)
void pool_dense_kernel(const float* __restrict__ h2, const float* __restrict__ att_w,
                       const float* __restrict__ d1W, const float* __restrict__ d1b,
                       const float* __restrict__ d2W, const float* __restrict__ d2b,
                       float* __restrict__ out)
{
    int b   = blockIdx.x;
    int tid = threadIdx.x;

    __shared__ float red[256];
    __shared__ float sP[Nn];        // softmax weights, 6 KB
    __shared__ float sRep[DATT];
    __shared__ float sZ[DH];

    const float* h2b = h2 + (size_t)b * Nn * DATT;

    // phase 1: tanh(h . att_w), track max
    float tv[(Nn + 255) / 256];
    float mx = -3e38f;
    int   it = 0;
    for (int n = tid; n < Nn; n += 256, ++it) {
        const float* hr = h2b + (size_t)n * DATT;
        float s = 0.f;
#pragma unroll 8
        for (int f = 0; f < DATT; ++f) s += hr[f] * att_w[f];
        float t = tanhf(s);
        tv[it] = t;
        mx = fmaxf(mx, t);
    }
    red[tid] = mx;
    __syncthreads();
    for (int off = 128; off > 0; off >>= 1) {
        if (tid < off) red[tid] = fmaxf(red[tid], red[tid + off]);
        __syncthreads();
    }
    mx = red[0];
    __syncthreads();

    // phase 2: exp weights into LDS + sum
    float ps = 0.f;
    it = 0;
    for (int n = tid; n < Nn; n += 256, ++it) {
        float p = __expf(tv[it] - mx);
        sP[n] = p;
        ps += p;
    }
    red[tid] = ps;
    __syncthreads();
    for (int off = 128; off > 0; off >>= 1) {
        if (tid < off) red[tid] += red[tid + off];
        __syncthreads();
    }
    float inv = 1.f / red[0];
    __syncthreads();

    // phase 3: pooled rep, thread f owns feature f (coalesced column reads)
    if (tid < DATT) {
        float acc = 0.f;
        for (int n = 0; n < Nn; ++n)
            acc += sP[n] * h2b[(size_t)n * DATT + tid];
        sRep[tid] = acc * inv;
    }
    __syncthreads();

    // phase 4: Dense(192 -> 128) + ReLU
    if (tid < DH) {
        float z = d1b[tid];
#pragma unroll 8
        for (int f = 0; f < DATT; ++f)
            z += sRep[f] * d1W[f * DH + tid];
        sZ[tid] = fmaxf(z, 0.f);
    }
    __syncthreads();

    // phase 5: Dense(128 -> 1)
    float v = (tid < DH) ? sZ[tid] * d2W[tid] : 0.f;
    red[tid] = v;
    __syncthreads();
    for (int off = 128; off > 0; off >>= 1) {
        if (tid < off) red[tid] += red[tid + off];
        __syncthreads();
    }
    if (tid == 0) out[b] = red[0] + d2b[0];
}

// ---------------------------------------------------------------------------
extern "C" void kernel_launch(void* const* d_in, const int* in_sizes, int n_in,
                              void* d_out, int out_size, void* d_ws, size_t ws_size,
                              hipStream_t stream)
{
    const float* V     = (const float*)d_in[0];
    const float* adj   = (const float*)d_in[1];
    const float* W1    = (const float*)d_in[2];
    const float* a1s   = (const float*)d_in[3];
    const float* a1d   = (const float*)d_in[4];
    const float* W2    = (const float*)d_in[5];
    const float* a2s   = (const float*)d_in[6];
    const float* a2d   = (const float*)d_in[7];
    const float* att_w = (const float*)d_in[8];
    const float* d1W   = (const float*)d_in[9];
    const float* d1b   = (const float*)d_in[10];
    const float* d2W   = (const float*)d_in[11];
    const float* d2b   = (const float*)d_in[12];
    float* outp = (float*)d_out;

    // workspace carve-up (256B aligned)
    char*  ws  = (char*)d_ws;
    size_t off = 0;
    auto carve = [&](size_t bytes) -> void* {
        void* p = ws + off;
        off = (off + bytes + 255) & ~(size_t)255;
        return p;
    };
    const size_t BHN = (size_t)Bsz * HH * Nn;

    _Float16* Wh1T = (_Float16*)carve((size_t)Bsz * HH * F1c * Np * sizeof(_Float16));
    float*    es1  = (float*)carve(BHN * sizeof(float));
    float*    ed1  = (float*)carve(BHN * sizeof(float));
    float*    h1   = (float*)carve((size_t)Bsz * Nn * (HH * F1c) * sizeof(float));
    _Float16* Wh2T = (_Float16*)carve((size_t)Bsz * HH * F2c * Np * sizeof(_Float16));
    float*    es2  = (float*)carve(BHN * sizeof(float));
    float*    ed2  = (float*)carve(BHN * sizeof(float));
    float*    h2   = (float*)carve((size_t)Bsz * Nn * DATT * sizeof(float));

    const int nNodes  = Bsz * Nn;
    const int RT      = (Nn + 15) / 16;
    const int tasks   = Bsz * HH * RT;
    const int attnBlk = (tasks + 7) / 8;

    // ---- layer 1 (single fused adj pass) ----
    feat_kernel<FIN1, F1c><<<(nNodes + 255) / 256, 256, 0, stream>>>(
        V, W1, a1s, a1d, Wh1T, es1, ed1);
    attn_kernel<F1c><<<attnBlk, 256, 0, stream>>>(adj, Wh1T, es1, ed1, h1);

    // ---- layer 2 (single fused adj pass) ----
    feat_kernel<HH * F1c, F2c><<<(nNodes + 255) / 256, 256, 0, stream>>>(
        h1, W2, a2s, a2d, Wh2T, es2, ed2);
    attn_kernel<F2c><<<attnBlk, 256, 0, stream>>>(adj, Wh2T, es2, ed2, h2);

    // ---- pooling + MLP head ----
    pool_dense_kernel<<<Bsz, 256, 0, stream>>>(h2, att_w, d1W, d1b, d2W, d2b, outp);
}
